// RelativePositionalAttention2_12146167513398
// MI455X (gfx1250) — matrix operand order repaired and checked by reading, not compiled, not run on hardware
//
#include <hip/hip_runtime.h>
#include <hip/hip_bf16.h>
#include <stdint.h>

// ---- problem constants (match reference) ----
#define BATCH   4
#define SEQ     1024
#define EMBED   512
#define HEADS   8
#define DH      64
#define TOKENS  (BATCH * SEQ)       // 4096
#define KDIM    512                 // inner dim of all three projections

typedef __attribute__((ext_vector_type(16))) _Float16 v16h;
typedef __attribute__((ext_vector_type(8)))  _Float16 v8h;
typedef __attribute__((ext_vector_type(8)))  float    v8f;
typedef __attribute__((ext_vector_type(4)))  unsigned int v4ui;
typedef __attribute__((ext_vector_type(8)))  int      v8i;
typedef __attribute__((ext_vector_type(4)))  int      v4i;

// D = A(16x32 f16) * B(32x16 f16) + C(16x16 f32)
__device__ __forceinline__ v8f wmma16(v16h a, v16h b, v8f c) {
  return __builtin_amdgcn_wmma_f32_16x16x32_f16(
      false, a, false, b, (short)0, c, false, false);
}

// build a v16h fragment from two 16B-aligned 8-half chunks
__device__ __forceinline__ v16h load2x8(const _Float16* p0, const _Float16* p1) {
  v8h lo = *(const v8h*)p0;
  v8h hi = *(const v8h*)p1;
  v16h r;
#pragma unroll
  for (int i = 0; i < 8; ++i) { r[i] = lo[i]; r[i + 8] = hi[i]; }
  return r;
}

// ---------------------------------------------------------------------------
// Tensor Data Mover: DMA a (tile_rows x row_elems) f16 tile (row-major,
// stride row_stride_elems) from global memory into LDS at lds_addr.
// D# layout per CDNA5 ISA:
//   g0: [1:0]=count(1), [63:32]=lds_addr, [120:64]=global_addr, [127:126]=2
//   g1: [17:16]=data_size(1 => 2B), [79:48]=tensor_dim0, [111:80]=tensor_dim1,
//       [127:112]=tile_dim0, [143:128]=tile_dim1, [207:160]=tensor_dim0_stride
// Groups 2/3 unused (2D tile) -> zero. 6-arg builtin (clang-23 lane).
// ---------------------------------------------------------------------------
__device__ __forceinline__ void tdm_load_tile_f16(
    unsigned lds_addr, const void* gptr,
    unsigned tile_rows, unsigned row_elems,
    unsigned row_stride_elems, unsigned tensor_rows) {
  unsigned long long ga = (unsigned long long)(uintptr_t)gptr;
  v4ui g0;
  g0[0] = 1u;                                              // count=1
  g0[1] = lds_addr;                                        // LDS byte address
  g0[2] = (unsigned)(ga & 0xFFFFFFFFu);                    // global_addr lo
  g0[3] = (unsigned)((ga >> 32) & 0x01FFFFFFu) | (2u << 30); // addr hi | type=2
  v8i g1;
  g1[0] = (int)(1u << 16);                                 // data_size=1 (2B)
  g1[1] = (int)((row_elems & 0xFFFFu) << 16);              // tensor_dim0 lo @48
  g1[2] = (int)(((row_elems >> 16) & 0xFFFFu) |
                ((tensor_rows & 0xFFFFu) << 16));          // dim0 hi | dim1 lo
  g1[3] = (int)(((tensor_rows >> 16) & 0xFFFFu) |
                ((row_elems & 0xFFFFu) << 16));            // dim1 hi | tile_dim0
  g1[4] = (int)(tile_rows & 0xFFFFu);                      // tile_dim1, dim2=0
  g1[5] = (int)row_stride_elems;                           // dim0_stride lo
  g1[6] = 0;
  g1[7] = 0;
  v4i z4 = (v4i)0;
  v8i z8 = (v8i)0;
  __builtin_amdgcn_tensor_load_to_lds(g0, g1, z4, z4, z8, 0);
}

// ---------------------------------------------------------------------------
// 1) f32 -> f16 conversion
// ---------------------------------------------------------------------------
__global__ void f32_to_f16_kernel(const float* __restrict__ in,
                                  _Float16* __restrict__ out, int n) {
  int i = blockIdx.x * blockDim.x + threadIdx.x;
  if (i < n) out[i] = (_Float16)in[i];
}

// ---------------------------------------------------------------------------
// 2) GEMM: C[m,n] = sum_k A[m,k] * Bw[n,k], K = 512, f16 in, f32 out.
//    Block = 256 threads (8 waves, 2x4 wave grid).
//    Block tile: 32 rows x 256 cols; wave tile: 16 rows x 64 cols (4 accums).
//    A strip (32 x 512 f16 = 32KB) DMA'd into LDS by the TDM.
// ---------------------------------------------------------------------------
__global__ __launch_bounds__(256) void gemm_xw_kernel(
    const _Float16* __restrict__ A, const _Float16* __restrict__ Bw,
    float* __restrict__ C, int Ncols) {
  __shared__ _Float16 As[32 * KDIM];        // 32 KB

  const int m0 = blockIdx.x * 32;

  if (threadIdx.x < 32) {                   // wave 0 issues the DMA
    tdm_load_tile_f16((unsigned)(uintptr_t)As,
                      A + (size_t)m0 * KDIM,
                      /*tile_rows=*/32, /*row_elems=*/KDIM,
                      /*row_stride=*/KDIM, /*tensor_rows=*/TOKENS);
    __builtin_amdgcn_s_wait_tensorcnt(0);
  }
  __syncthreads();

  const int wave = threadIdx.x >> 5, lane = threadIdx.x & 31;
  const int lo = lane & 15, hi = lane >> 4;
  const int wr = wave >> 2;                 // 0..1 : row group
  const int wc = wave & 3;                  // 0..3 : col group
  const int n0 = blockIdx.y * 256 + wc * 64;

  const _Float16* arow = &As[(wr * 16 + lo) * KDIM];
  const _Float16* bbase = Bw + (size_t)(n0 + lo) * KDIM;

  v8f acc[4];
#pragma unroll
  for (int c = 0; c < 4; ++c) acc[c] = (v8f)0.0f;

#pragma unroll 2
  for (int k0 = 0; k0 < KDIM; k0 += 32) {
    v16h a = load2x8(arow + k0 + 16 * hi, arow + k0 + 16 * hi + 8);
#pragma unroll
    for (int c = 0; c < 4; ++c) {
      const _Float16* brow = bbase + (size_t)(c * 16) * KDIM;
      __builtin_prefetch(brow + k0 + 128, 0, 0);       // global_prefetch_b8
      v16h b = load2x8(brow + k0 + 16 * hi, brow + k0 + 16 * hi + 8);
      acc[c] = wmma16(a, b, acc[c]);
    }
  }

#pragma unroll
  for (int c = 0; c < 4; ++c) {
#pragma unroll
    for (int r = 0; r < 8; ++r) {
      int row = m0 + wr * 16 + r + 8 * hi;
      int col = n0 + c * 16 + lo;
      C[(size_t)row * Ncols + col] = acc[c][r];
    }
  }
}

// ---------------------------------------------------------------------------
// 3) RoPE + layout shuffle.
//    qraw [4096,512] f32, kvraw [4096,1024] f32 (k | v).
//    Out: Qh,Kh f16 [B,H,N,DH]; Vt f16 [B,H,DH,N] (transposed for PV WMMA).
// ---------------------------------------------------------------------------
__global__ __launch_bounds__(256) void rope_kernel(
    const float* __restrict__ qraw, const float* __restrict__ kvraw,
    const float* __restrict__ coords,
    const float* __restrict__ invt, const float* __restrict__ invx,
    const float* __restrict__ invy,
    _Float16* __restrict__ Qh, _Float16* __restrict__ Kh,
    _Float16* __restrict__ Vt) {
  int idx = blockIdx.x * 256 + threadIdx.x;   // token*HEADS + h
  int m = idx >> 3;
  int h = idx & 7;
  int b = m >> 10, n = m & (SEQ - 1);

  float ct = coords[m * 3 + 0];
  float cx = coords[m * 3 + 1];
  float cy = coords[m * 3 + 2];

  const float* q = qraw  + (size_t)m * EMBED + h * DH;
  const float* k = kvraw + (size_t)m * (2 * EMBED) + h * DH;
  const float* v = kvraw + (size_t)m * (2 * EMBED) + EMBED + h * DH;

  size_t bh = (size_t)b * HEADS + h;
  _Float16* qo = Qh + (bh * SEQ + n) * DH;
  _Float16* ko = Kh + (bh * SEQ + n) * DH;

#pragma unroll
  for (int i = 0; i < 32; ++i) {            // 32 rotation pairs
    float coord, inv;
    if (i < 8)       { coord = ct; inv = invt[i]; }       // ROT 16 -> 8 pairs
    else if (i < 20) { coord = cx; inv = invx[i - 8]; }   // ROT 24 -> 12 pairs
    else             { coord = cy; inv = invy[i - 20]; }  // ROT 24 -> 12 pairs
    float f = coord * inv;
    float c = __cosf(f), s = __sinf(f);
    float q0 = q[2 * i], q1 = q[2 * i + 1];
    float k0 = k[2 * i], k1 = k[2 * i + 1];
    qo[2 * i]     = (_Float16)(q0 * c - q1 * s);
    qo[2 * i + 1] = (_Float16)(q1 * c + q0 * s);
    ko[2 * i]     = (_Float16)(k0 * c - k1 * s);
    ko[2 * i + 1] = (_Float16)(k1 * c + k0 * s);
  }
#pragma unroll
  for (int d = 0; d < DH; ++d)
    Vt[(bh * DH + d) * SEQ + n] = (_Float16)v[d];
}

// ---------------------------------------------------------------------------
// 4) Flash attention. grid = (B*H, SEQ/128), block = 256 (8 waves).
//    Each wave: 16 queries, streams 32 keys/iter; 256 dynamic wmma/wave.
// ---------------------------------------------------------------------------
__global__ __launch_bounds__(256) void attn_kernel(
    const _Float16* __restrict__ Qh, const _Float16* __restrict__ Kh,
    const _Float16* __restrict__ Vt, const unsigned char* __restrict__ pad,
    _Float16* __restrict__ Oh) {
  __shared__ _Float16 P[8][16 * 32];        // per-wave P staging

  const int bh = blockIdx.x;                // 0..31
  const int b = bh >> 3, h = bh & 7;
  const int wave = threadIdx.x >> 5, lane = threadIdx.x & 31;
  const int lo = lane & 15, hi = lane >> 4;
  const int qm0 = blockIdx.y * 128 + wave * 16;   // query tile in [0,1024)

  const _Float16* Qb = Qh + (size_t)bh * SEQ * DH;
  const _Float16* Kb = Kh + (size_t)bh * SEQ * DH;
  const _Float16* Vb = Vt + (size_t)bh * DH * SEQ;
  const unsigned char* padb = pad + (size_t)b * SEQ;

  // Q fragments (A layout) for dh chunks [0,32) and [32,64)
  const _Float16* qrow = Qb + (size_t)(qm0 + lo) * DH;
  v16h aq0 = load2x8(qrow + 16 * hi,      qrow + 16 * hi + 8);
  v16h aq1 = load2x8(qrow + 32 + 16 * hi, qrow + 40 + 16 * hi);

  bool qinv[8];
#pragma unroll
  for (int r = 0; r < 8; ++r) qinv[r] = padb[qm0 + r + 8 * hi] != 0;

  v8f o[4];
#pragma unroll
  for (int c = 0; c < 4; ++c) o[c] = (v8f)0.0f;
  float rmax[8], rsum[8];
#pragma unroll
  for (int r = 0; r < 8; ++r) { rmax[r] = -3.0e38f; rsum[r] = 0.0f; }

  _Float16* Pw = P[wave];
  const v8f z = (v8f)0.0f;

  for (int kb = 0; kb < SEQ; kb += 32) {
    float sc[2][8];
#pragma unroll
    for (int t = 0; t < 2; ++t) {
      const int key = kb + t * 16 + lo;
      const _Float16* krow = Kb + (size_t)key * DH;
      v16h bk0 = load2x8(krow + 16 * hi,      krow + 16 * hi + 8);
      v16h bk1 = load2x8(krow + 32 + 16 * hi, krow + 40 + 16 * hi);
      v8f s = wmma16(aq0, bk0, z);
      s = wmma16(aq1, bk1, s);
      const bool kinvd = padb[key] != 0;
#pragma unroll
      for (int r = 0; r < 8; ++r) {
        float sv = s[r] * 0.125f;                 // 1/sqrt(DH)
        if (kinvd || qinv[r]) sv = -1.0e9f;       // same constant as reference
        sc[t][r] = sv;
      }
    }
    // online softmax: per-row stats live across lanes 0..15 of each half
#pragma unroll
    for (int r = 0; r < 8; ++r) {
      float t = fmaxf(sc[0][r], sc[1][r]);
      t = fmaxf(t, __shfl_xor(t, 1, 32));
      t = fmaxf(t, __shfl_xor(t, 2, 32));
      t = fmaxf(t, __shfl_xor(t, 4, 32));
      t = fmaxf(t, __shfl_xor(t, 8, 32));
      float nm    = fmaxf(rmax[r], t);
      float scale = __expf(rmax[r] - nm);
      float e0 = __expf(sc[0][r] - nm);
      float e1 = __expf(sc[1][r] - nm);
      float rs = e0 + e1;
      rs += __shfl_xor(rs, 1, 32);
      rs += __shfl_xor(rs, 2, 32);
      rs += __shfl_xor(rs, 4, 32);
      rs += __shfl_xor(rs, 8, 32);
      rsum[r] = rsum[r] * scale + rs;
      rmax[r] = nm;
#pragma unroll
      for (int c = 0; c < 4; ++c) o[c][r] *= scale;
      Pw[(r + 8 * hi) * 32 + lo]      = (_Float16)e0;
      Pw[(r + 8 * hi) * 32 + 16 + lo] = (_Float16)e1;
    }
    asm volatile("s_wait_dscnt 0" ::: "memory");   // cross-lane LDS RAW

    // P as A fragment (16 queries x 32 keys)
    const _Float16* prow = Pw + lo * 32;
    v16h ap = load2x8(prow + 16 * hi, prow + 16 * hi + 8);

    // O += P * V   (V transposed: row = dh, contiguous in key)
#pragma unroll
    for (int c = 0; c < 4; ++c) {
      const _Float16* vrow = Vb + (size_t)(c * 16 + lo) * SEQ + kb + 16 * hi;
      v16h bv = load2x8(vrow, vrow + 8);
      o[c] = wmma16(ap, bv, o[c]);
    }
  }

  // normalize and store to Oh [tokens, EMBED] f16
#pragma unroll
  for (int r = 0; r < 8; ++r) {
    float invl = 1.0f / rsum[r];
    size_t base = (size_t)(b * SEQ + qm0 + r + 8 * hi) * EMBED + h * DH + lo;
    Oh[base]      = (_Float16)(o[0][r] * invl);
    Oh[base + 16] = (_Float16)(o[1][r] * invl);
    Oh[base + 32] = (_Float16)(o[2][r] * invl);
    Oh[base + 48] = (_Float16)(o[3][r] * invl);
  }
}

// ---------------------------------------------------------------------------
// launch
// ---------------------------------------------------------------------------
extern "C" void kernel_launch(void* const* d_in, const int* in_sizes, int n_in,
                              void* d_out, int out_size, void* d_ws,
                              size_t ws_size, hipStream_t stream) {
  const float* x      = (const float*)d_in[0];
  const float* coords = (const float*)d_in[1];
  const float* Wq     = (const float*)d_in[2];
  const float* Wkv    = (const float*)d_in[3];
  const float* Wo     = (const float*)d_in[4];
  const float* invt   = (const float*)d_in[5];
  const float* invx   = (const float*)d_in[6];
  const float* invy   = (const float*)d_in[7];
  const unsigned char* pad = (const unsigned char*)d_in[8];
  float* out = (float*)d_out;
  (void)in_sizes; (void)n_in; (void)out_size; (void)ws_size;

  char* ws = (char*)d_ws;
  size_t off = 0;
  auto alloc = [&](size_t bytes) -> char* {
    char* p = ws + off;
    off = (off + bytes + 255) & ~(size_t)255;
    return p;
  };

  _Float16* xh   = (_Float16*)alloc((size_t)TOKENS * EMBED * 2);
  _Float16* wqh  = (_Float16*)alloc((size_t)EMBED * EMBED * 2);
  _Float16* wkvh = (_Float16*)alloc((size_t)2 * EMBED * EMBED * 2);
  _Float16* woh  = (_Float16*)alloc((size_t)EMBED * EMBED * 2);
  float*    qraw = (float*)alloc((size_t)TOKENS * EMBED * 4);
  float*    kvraw= (float*)alloc((size_t)TOKENS * 2 * EMBED * 4);
  _Float16* Qh   = (_Float16*)alloc((size_t)TOKENS * EMBED * 2);
  _Float16* Kh   = (_Float16*)alloc((size_t)TOKENS * EMBED * 2);
  _Float16* Vt   = (_Float16*)alloc((size_t)TOKENS * EMBED * 2);
  _Float16* Oh   = (_Float16*)alloc((size_t)TOKENS * EMBED * 2);

  const int nx = TOKENS * EMBED;          // 2,097,152
  const int nw = EMBED * EMBED;           // 262,144
  f32_to_f16_kernel<<<(nx + 255) / 256, 256, 0, stream>>>(x, xh, nx);
  f32_to_f16_kernel<<<(nw + 255) / 256, 256, 0, stream>>>(Wq, wqh, nw);
  f32_to_f16_kernel<<<(2 * nw + 255) / 256, 256, 0, stream>>>(Wkv, wkvh, 2 * nw);
  f32_to_f16_kernel<<<(nw + 255) / 256, 256, 0, stream>>>(Wo, woh, nw);

  // Q = X * Wq^T ; KV = X * Wkv^T   (32-row x 256-col block tiles)
  gemm_xw_kernel<<<dim3(TOKENS / 32, EMBED / 256), 256, 0, stream>>>(
      xh, wqh, qraw, EMBED);
  gemm_xw_kernel<<<dim3(TOKENS / 32, (2 * EMBED) / 256), 256, 0, stream>>>(
      xh, wkvh, kvraw, 2 * EMBED);

  // RoPE + f16 repack (V transposed)
  rope_kernel<<<(TOKENS * HEADS) / 256, 256, 0, stream>>>(
      qraw, kvraw, coords, invt, invx, invy, Qh, Kh, Vt);

  // attention
  attn_kernel<<<dim3(BATCH * HEADS, SEQ / 128), 256, 0, stream>>>(
      Qh, Kh, Vt, pad, Oh);

  // out = O * Wo^T (f32 result straight into d_out)
  gemm_xw_kernel<<<dim3(TOKENS / 32, EMBED / 256), 256, 0, stream>>>(
      Oh, woh, out, EMBED);
}
